// GraphSage_27788438405728
// MI455X (gfx1250) — compile-verified, matrix operand order checked by reading, not run on hardware
//
#include <hip/hip_runtime.h>
#include <hip/hip_bf16.h>

typedef __attribute__((ext_vector_type(16))) __bf16 v16bf;
typedef __attribute__((ext_vector_type(8)))  __bf16 v8bf;
typedef __attribute__((ext_vector_type(4)))  __bf16 v4bf;
typedef __attribute__((ext_vector_type(8)))  float  v8f;

#define BATCH 4096
#define KN    64
#define D     128
#define HLD   136   /* 128 + 8 pad (bf16): row stride 272B -> bank step 4, conflict-free b128 */

// d_ws layout: wT1 (128 x 256 bf16, row-major, = w1 transposed)  @ 0
//              wT2 (128 x 128 bf16, row-major, = w2 transposed)  @ 65536
#define WT1_OFF 0
#define WT2_OFF 65536

// ---- prologue: transpose + convert weights f32 -> bf16 into workspace ----
__global__ __launch_bounds__(256)
void convert_weights(const float* __restrict__ w1, const float* __restrict__ w2,
                     __bf16* __restrict__ wT1, __bf16* __restrict__ wT2)
{
    const int id = blockIdx.x * 256 + threadIdx.x;      // 192 blocks -> 49152 threads
    if (id < 256 * 128) {
        const int n = id >> 8, k = id & 255;            // wT1[n][k] = w1[k][n]
        wT1[id] = (__bf16)w1[k * D + n];
    } else {
        const int j = id - 256 * 128;                   // 16384 elems
        const int n = j >> 7, k = j & 127;              // wT2[n][k] = w2[k][n]
        wT2[j] = (__bf16)w2[k * D + n];
    }
}

__global__ __launch_bounds__(256)
void graphsage_fused(const int* __restrict__ node_idx,
                     const int* __restrict__ neigh_idx,
                     const int* __restrict__ neigh_len,
                     const float* __restrict__ u2e_visual,
                     const float* __restrict__ u2e_text,
                     const __bf16* __restrict__ wT1,   // [128][256] bf16 (w1^T)
                     const float* __restrict__ b1,
                     const __bf16* __restrict__ wT2,   // [128][128] bf16 (w2^T)
                     const float* __restrict__ b2,
                     const float* __restrict__ w3, const float* __restrict__ b3,
                     float* __restrict__ out)
{
    __shared__ __align__(16) __bf16 sx[KN][HLD];   // gathered neighbor features, bf16
    __shared__ __align__(16) __bf16 sh1[KN][HLD];  // hidden layer 1, bf16
    __shared__ __align__(16) float  su[D];         // node rep, f32 (exact passthrough)
    __shared__ float ssc[KN];                      // scores -> attention weights
    __shared__ int   snb[KN];

    const int b    = blockIdx.x;
    const int mod  = blockIdx.y;           // 0 = visual, 1 = text
    const int tid  = threadIdx.x;
    const int lane = tid & 31;
    const int wv   = tid >> 5;             // 8 waves, wave -> 16-col N tile
    const int col  = lane & 15;
    const int hil  = (lane >> 4) & 1;      // lane half selects K sub-block
    const int n    = wv * 16 + col;        // this lane's output column 0..127

    const float* table = mod ? u2e_text : u2e_visual;
    const int len = neigh_len[b];

    const __bf16* w1r = wT1 + n * 256;     // this lane's w1 column (transposed row)
    const __bf16* w2r = wT2 + n * 128;

    // ---- phase 0: indices, node row, score init (scores start at b3) ----
    if (tid < KN) { snb[tid] = neigh_idx[b * KN + tid]; ssc[tid] = b3[0]; }
    if (tid < 32) {
        const float4 v = ((const float4*)(table + (size_t)node_idx[b] * D))[tid];
        ((float4*)su)[tid] = v;
    }
    __syncthreads();

    // ---- phase 1: gather 64 neighbor rows (f32 -> bf16) ----
    for (int u = tid; u < KN * 32; u += 256) {
        const int row = u >> 5, seg = u & 31;
        const float4 v = ((const float4*)(table + (size_t)snb[row] * D))[seg];
        v4bf pv; pv[0] = (__bf16)v.x; pv[1] = (__bf16)v.y; pv[2] = (__bf16)v.z; pv[3] = (__bf16)v.w;
        *(v4bf*)&sx[row][seg * 4] = pv;
    }
    __syncthreads();

    // ---- node matvec: nvc = b1[n] + sum_k node[k] * w1[128+k][n]  (rank-1 part of GEMM1)
    float mv = 0.f;
    {
        const int kbase = hil * 64;        // lanes 0-15: k 0..63, lanes 16-31: k 64..127
        #pragma unroll
        for (int c = 0; c < 8; ++c) {
            const int kk = kbase + c * 8;
            const v8bf wvv = *(const v8bf*)&w1r[128 + kk];
            const float4 sa = *(const float4*)&su[kk];
            const float4 sb = *(const float4*)&su[kk + 4];
            mv += sa.x * (float)wvv[0] + sa.y * (float)wvv[1]
                + sa.z * (float)wvv[2] + sa.w * (float)wvv[3]
                + sb.x * (float)wvv[4] + sb.y * (float)wvv[5]
                + sb.z * (float)wvv[6] + sb.w * (float)wvv[7];
        }
        mv += __shfl_xor(mv, 16, 32);      // combine the two k-halves
    }
    const float nvc = b1[n] + mv;

    // ---- GEMM1 (neighbor part): h1 = relu(neigh @ w1a + nvc), M=64 N=128 K=128 ----
    v8f acc[4];
    #pragma unroll
    for (int mt = 0; mt < 4; ++mt) acc[mt] = (v8f){};

    #pragma unroll
    for (int k0 = 0; k0 < 128; k0 += 32) {
        const int kb = k0 + hil * 8;
        const v8bf blo = *(const v8bf*)&w1r[kb];
        const v8bf bhi = *(const v8bf*)&w1r[kb + 16];
        v16bf bf;
        #pragma unroll
        for (int j = 0; j < 8; ++j) { bf[j] = blo[j]; bf[j + 8] = bhi[j]; }
        #pragma unroll
        for (int mt = 0; mt < 4; ++mt) {
            const int row = mt * 16 + col;
            const v8bf lo = *(const v8bf*)&sx[row][kb];
            const v8bf hh = *(const v8bf*)&sx[row][kb + 16];
            v16bf af;
            #pragma unroll
            for (int j = 0; j < 8; ++j) { af[j] = lo[j]; af[j + 8] = hh[j]; }
            acc[mt] = __builtin_amdgcn_wmma_f32_16x16x32_bf16(
                false, af, false, bf, (short)0, acc[mt], false, false);
        }
    }

    {   // bias(+node term) + relu + store h1 (bf16)
        #pragma unroll
        for (int mt = 0; mt < 4; ++mt) {
            #pragma unroll
            for (int i = 0; i < 8; ++i) {
                const int row = mt * 16 + hil * 8 + i;
                float v = acc[mt][i] + nvc;
                v = v > 0.f ? v : 0.f;
                sh1[row][n] = (__bf16)v;   // C layout: N = n for both lane halves
            }
        }
    }
    __syncthreads();

    // ---- GEMM2: h2 = relu(h1 @ w2 + b2); fold scores = h2 @ w3 into epilogue ----
    #pragma unroll
    for (int mt = 0; mt < 4; ++mt) acc[mt] = (v8f){};

    #pragma unroll
    for (int k0 = 0; k0 < 128; k0 += 32) {
        const int kb = k0 + hil * 8;
        const v8bf blo = *(const v8bf*)&w2r[kb];
        const v8bf bhi = *(const v8bf*)&w2r[kb + 16];
        v16bf bf;
        #pragma unroll
        for (int j = 0; j < 8; ++j) { bf[j] = blo[j]; bf[j + 8] = bhi[j]; }
        #pragma unroll
        for (int mt = 0; mt < 4; ++mt) {
            const int row = mt * 16 + col;
            const v8bf lo = *(const v8bf*)&sh1[row][kb];
            const v8bf hh = *(const v8bf*)&sh1[row][kb + 16];
            v16bf af;
            #pragma unroll
            for (int j = 0; j < 8; ++j) { af[j] = lo[j]; af[j + 8] = hh[j]; }
            acc[mt] = __builtin_amdgcn_wmma_f32_16x16x32_bf16(
                false, af, false, bf, (short)0, acc[mt], false, false);
        }
    }

    {   // score partials: relu(h2)*w3, reduce across the 16-lane column group
        const float b2c = b2[n];
        const float w3c = w3[n];
        #pragma unroll
        for (int mt = 0; mt < 4; ++mt) {
            float v[8];
            #pragma unroll
            for (int i = 0; i < 8; ++i) {
                float t = acc[mt][i] + b2c;
                t = t > 0.f ? t : 0.f;
                v[i] = t * w3c;
            }
            #pragma unroll
            for (int s = 1; s < 16; s <<= 1)
                #pragma unroll
                for (int i = 0; i < 8; ++i) v[i] += __shfl_xor(v[i], s, 32);
            if (col == 0) {                      // lane 0 -> rows mt*16+i, lane 16 -> +8
                const int base = mt * 16 + hil * 8;
                #pragma unroll
                for (int i = 0; i < 8; ++i) atomicAdd(&ssc[base + i], v[i]);
            }
        }
    }
    __syncthreads();

    // ---- masked softmax over K=64 (wave 0, 2 elems per lane) ----
    if (tid < 32) {
        float s0 = (tid      < len) ? ssc[tid]      : -1e30f;
        float s1 = (tid + 32 < len) ? ssc[tid + 32] : -1e30f;
        float m = fmaxf(s0, s1);
        #pragma unroll
        for (int s = 1; s < 32; s <<= 1) m = fmaxf(m, __shfl_xor(m, s, 32));
        const float e0 = __expf(s0 - m), e1 = __expf(s1 - m);
        float sum = e0 + e1;
        #pragma unroll
        for (int s = 1; s < 32; s <<= 1) sum += __shfl_xor(sum, s, 32);
        const float inv = 1.0f / sum;
        ssc[tid] = e0 * inv;
        ssc[tid + 32] = e1 * inv;
    }
    __syncthreads();

    // ---- aggregate + write: plane mod = node rep, plane 2+mod = attention agg ----
    if (tid < D) {
        const float uval = su[tid];
        float agg;
        if (len > 0) {
            agg = 0.f;
            #pragma unroll 4
            for (int k = 0; k < KN; ++k) agg += ssc[k] * (float)sx[k][tid];
        } else {
            agg = uval;   // empty neighbor list -> self representation
        }
        const size_t plane = (size_t)BATCH * D;
        const size_t o = (size_t)b * D + tid;
        out[(size_t)mod       * plane + o] = uval;
        out[(size_t)(2 + mod) * plane + o] = agg;
    }
}

extern "C" void kernel_launch(void* const* d_in, const int* in_sizes, int n_in,
                              void* d_out, int out_size, void* d_ws, size_t ws_size,
                              hipStream_t stream) {
    const int*   node_idx  = (const int*)d_in[0];
    const int*   neigh_idx = (const int*)d_in[1];
    const int*   neigh_len = (const int*)d_in[2];
    const float* u2e_vis   = (const float*)d_in[3];
    const float* u2e_txt   = (const float*)d_in[4];
    const float* w1 = (const float*)d_in[5];
    const float* b1 = (const float*)d_in[6];
    const float* w2 = (const float*)d_in[7];
    const float* b2 = (const float*)d_in[8];
    const float* w3 = (const float*)d_in[9];
    const float* b3 = (const float*)d_in[10];
    float* out = (float*)d_out;

    __bf16* wT1 = (__bf16*)((char*)d_ws + WT1_OFF);
    __bf16* wT2 = (__bf16*)((char*)d_ws + WT2_OFF);

    // 1) transpose + convert weights once (bf16, [N][K] layout for WMMA B-fragments)
    convert_weights<<<192, 256, 0, stream>>>(w1, w2, wT1, wT2);

    // 2) fused gather + 2x WMMA GEMM + attention + aggregate
    dim3 grid(BATCH, 2);
    graphsage_fused<<<grid, 256, 0, stream>>>(node_idx, neigh_idx, neigh_len,
                                              u2e_vis, u2e_txt,
                                              wT1, b1, wT2, b2, w3, b3, out);
}
